// EdgeBiasedGATLayer_46815143526789
// MI455X (gfx1250) — matrix operand order repaired and checked by reading, not compiled, not run on hardware
//
#include <hip/hip_runtime.h>
#include <hip/hip_bf16.h>

typedef __bf16 bf16_t;
typedef __attribute__((ext_vector_type(16))) __bf16 v16bf;
typedef __attribute__((ext_vector_type(8)))  float  v8f;

#define HID   256
#define EDIM  16
#define NHEAD 8
#define HDIM  32
#define BSZ   4
#define NND   512
#define BN    (BSZ * NND)   // 2048 rows

union V16U {
  v16bf v;
  uint4 q[2];
};
union PK4 {
  bf16_t h[4];
  uint2 u;
};

// ---------------- WMMA fragment helpers (wave32, 16x16x32 bf16) ----------------
// A fragment from row-major [m][k] storage: two contiguous 16B chunks per lane.
// lane m = l&15, koff = (l&16); elems 0..15 -> k=koff..koff+15
__device__ __forceinline__ v16bf a_frag_g(const bf16_t* base, int stride, int lane) {
  V16U u;
  const bf16_t* p = base + (size_t)(lane & 15) * stride + (lane & 16);
  u.q[0] = *(const uint4*)(p);
  u.q[1] = *(const uint4*)(p + 8);
  return u.v;
}
// B fragment from fragment-packed tile (512 bf16 per 32x16 tile, lane-major):
// value at tile[lane*16 + i] == B[(lane&16)+i][lane&15]
__device__ __forceinline__ v16bf b_frag_packed(const bf16_t* tile, int lane) {
  V16U u;
  const uint4* p = (const uint4*)(tile + lane * 16);
  u.q[0] = p[0];
  u.q[1] = p[1];
  return u.v;
}
// B fragment where K runs contiguously in memory at fixed N (column-major source):
// per-lane 16 contiguous bf16
__device__ __forceinline__ v16bf b_frag_kcontig(const bf16_t* p) {
  V16U u;
  u.q[0] = *(const uint4*)(p);
  u.q[1] = *(const uint4*)(p + 8);
  return u.v;
}
__device__ __forceinline__ v8f wmma_bf16(v16bf a, v16bf b, v8f c) {
  return __builtin_amdgcn_wmma_f32_16x16x32_bf16(false, a, false, b, (short)0, c,
                                                 false, false);
}
// C/D (16x16 f32): lane n = l&15; VGPR r -> row m = r + ((l&16)>>1)

// ---------------- packing kernels ----------------
__global__ void k_f32_to_bf16_v4(const float* __restrict__ src, bf16_t* __restrict__ dst,
                                 int n4) {
  int i = blockIdx.x * blockDim.x + threadIdx.x;
  if (i < n4) {
    float4 f = *(const float4*)(src + (size_t)i * 4);
    PK4 pk;
    pk.h[0] = (bf16_t)f.x; pk.h[1] = (bf16_t)f.y;
    pk.h[2] = (bf16_t)f.z; pk.h[3] = (bf16_t)f.w;
    *(uint2*)(dst + (size_t)i * 4) = pk.u;
  }
}
// Pack W[srcK, srcN] (row-major fp32) into fragment order:
// dst[((kt*Ntiles + nt)*32 + lane)*16 + i] = W[kt*32 + (lane&16) + i][nt*16 + (lane&15)]
// zero-padded outside srcK/srcN.
__global__ void k_pack_frag(const float* __restrict__ W, bf16_t* __restrict__ dst,
                            int Ntiles, int srcK, int srcN, int total) {
  int e = blockIdx.x * blockDim.x + threadIdx.x;
  if (e >= total) return;
  const int i = e & 15;
  const int lane = (e >> 4) & 31;
  const int tile = e >> 9;
  const int nt = tile % Ntiles;
  const int kt = tile / Ntiles;
  const int k = kt * 32 + (lane & 16) + i;
  const int n = nt * 16 + (lane & 15);
  const float v = (k < srcK && n < srcN) ? W[(size_t)k * srcN + n] : 0.f;
  dst[e] = (bf16_t)v;
}

// ---------------- bf16 GEMM: Y = X[2048,256] @ W + bias ----------------
// mode 0: Y row-major [2048,256] bf16.  mode 1: Y = V^T layout [B,H,D,N] bf16.
__global__ __launch_bounds__(32) void k_gemm_bf16(const bf16_t* __restrict__ X,
                                                  const bf16_t* __restrict__ Wf,
                                                  const float* __restrict__ bias,
                                                  bf16_t* __restrict__ Y, int mode) {
  const int lane = threadIdx.x;
  const int ntile = blockIdx.x;          // 16 tiles of N
  const int m0 = blockIdx.y * 16;        // 128 tiles of M
  const int n = lane & 15;
  const int mh = (lane & 16) >> 1;
  v8f acc = {};
#pragma unroll
  for (int kt = 0; kt < 8; ++kt) {
    v16bf a = a_frag_g(X + (size_t)m0 * HID + kt * 32, HID, lane);
    v16bf b = b_frag_packed(Wf + (size_t)(kt * 16 + ntile) * 512, lane);
    acc = wmma_bf16(a, b, acc);
  }
  const int n0 = ntile * 16;
#pragma unroll
  for (int r = 0; r < 8; ++r) {
    const float v = acc[r] + bias[n0 + n];
    const int row = m0 + r + mh, col = n0 + n;
    if (mode == 0) {
      Y[(size_t)row * HID + col] = (bf16_t)v;
    } else {  // V^T: [B][H][D][N]
      const int bb = row >> 9, j = row & (NND - 1);
      const int hh = col >> 5, d = col & (HDIM - 1);
      Y[(((size_t)bb * NHEAD + hh) * HDIM + d) * NND + j] = (bf16_t)v;
    }
  }
}

// ---------------- fused edge MLP -> ebT[B,H,N,N] ----------------
__global__ __launch_bounds__(32) void k_edge_mlp(const float* __restrict__ edge_x,
                                                 const bf16_t* __restrict__ We1f,
                                                 const bf16_t* __restrict__ We2f,
                                                 const float* __restrict__ be1,
                                                 const float* __restrict__ be2,
                                                 float* __restrict__ ebT) {
  __shared__ __align__(16) bf16_t sH[16 * HID];  // hidden bf16, 8 KB
  const int lane = threadIdx.x;
  const int row0 = blockIdx.x * 16;
  const int n = lane & 15;
  const int mh = (lane & 16) >> 1;

  // A fragment in registers: real K=16, upper half zero (We1f also zero there)
  v16bf aE;
  {
    const float* p = edge_x + (size_t)(row0 + (lane & 15)) * EDIM;
    const float4 f0 = *(const float4*)(p);
    const float4 f1 = *(const float4*)(p + 4);
    aE[0] = (bf16_t)f0.x; aE[1] = (bf16_t)f0.y; aE[2] = (bf16_t)f0.z; aE[3] = (bf16_t)f0.w;
    aE[4] = (bf16_t)f1.x; aE[5] = (bf16_t)f1.y; aE[6] = (bf16_t)f1.z; aE[7] = (bf16_t)f1.w;
#pragma unroll
    for (int i = 8; i < 16; ++i) aE[i] = (lane & 16) ? (bf16_t)0.0f : (bf16_t)p[i];
#pragma unroll
    for (int i = 0; i < 8; ++i) if (lane & 16) aE[i] = (bf16_t)0.0f;
  }

  // GEMM1 + ReLU -> sH (bf16)
#pragma unroll
  for (int nt = 0; nt < 16; ++nt) {
    v8f c = {};
    c = wmma_bf16(aE, b_frag_packed(We1f + nt * 512, lane), c);
#pragma unroll
    for (int r = 0; r < 8; ++r) {
      float v = c[r] + be1[nt * 16 + n];
      sH[(r + mh) * HID + nt * 16 + n] = (bf16_t)(v > 0.f ? v : 0.f);
    }
  }
  __syncthreads();

  // GEMM2: [16,256] @ [256,16]
  v8f acc = {};
#pragma unroll
  for (int kt = 0; kt < 8; ++kt)
    acc = wmma_bf16(a_frag_g(&sH[kt * 32], HID, lane),
                    b_frag_packed(We2f + kt * 512, lane), acc);

  if (n < NHEAD) {
#pragma unroll
    for (int r = 0; r < 8; ++r) {
      const int gr = row0 + r + mh;          // flat (b,i,j)
      const int b_ = gr >> 18;               // N*N = 2^18
      const int rem = gr & (NND * NND - 1);
      const int i = rem >> 9, j = rem & (NND - 1);
      ebT[(((size_t)b_ * NHEAD + n) * NND + i) * NND + j] = acc[r] + be2[n];
    }
  }
}

// ---------------- attention: scores + softmax + attn@V ----------------
__global__ __launch_bounds__(32) void k_attention(const bf16_t* __restrict__ q_bf,
                                                  const bf16_t* __restrict__ kb_bf,
                                                  const bf16_t* __restrict__ vT,
                                                  const float* __restrict__ ebT,
                                                  const int* __restrict__ edge_mask,
                                                  float* __restrict__ attn_out,
                                                  bf16_t* __restrict__ ctx_bf) {
  __shared__ float  sS[16 * NND];   // 32 KB score strip
  __shared__ __align__(16) bf16_t sP[16 * NND];   // 16 KB bf16 probabilities
  const int lane = threadIdx.x;
  const int i0 = blockIdx.x * 16;
  const int h = blockIdx.y, b = blockIdx.z;
  const int n = lane & 15;
  const int mh = (lane & 16) >> 1;

  const v16bf qf = a_frag_g(q_bf + (size_t)(b * NND + i0) * HID + h * HDIM, HID, lane);

  for (int jt = 0; jt < 32; ++jt) {
    const int j0 = jt * 16;
    // B[k][n] = K[j0+n][d]: 16 contiguous bf16 per lane
    const bf16_t* kp =
        kb_bf + (size_t)(b * NND + j0 + n) * HID + h * HDIM + (lane & 16);
    v8f c = {};
    c = wmma_bf16(qf, b_frag_kcontig(kp), c);
#pragma unroll
    for (int r = 0; r < 8; ++r) {
      const int gi = i0 + r + mh, gj = j0 + n;
      float s = c[r] * 0.1767766952966369f;       // 1/sqrt(32)
      s = s > 0.f ? s : 0.2f * s;                 // leaky relu
      s += ebT[(((size_t)b * NHEAD + h) * NND + gi) * NND + gj];
      const bool valid = (edge_mask[((size_t)b * NND + gi) * NND + gj] != 0) || (gi == gj);
      sS[(r + mh) * NND + gj] = valid ? s : -1e9f;
    }
  }
  __syncthreads();

  // pair-lane softmax: lanes (l, l+16) share row l&15, each owns a contiguous 256-col half
  {
    const int row = lane & 15;
    const int jb = (lane & 16) ? 256 : 0;
    float* srow = &sS[row * NND + jb];
    float mx = -1e30f;
    for (int t = 0; t < 256; t += 4) {
      const float4 x = *(const float4*)&srow[t];
      mx = fmaxf(mx, fmaxf(fmaxf(x.x, x.y), fmaxf(x.z, x.w)));
    }
    mx = fmaxf(mx, __shfl_xor(mx, 16, 32));
    float sum = 0.f;
    for (int t = 0; t < 256; t += 4) {
      float4 x = *(const float4*)&srow[t];
      x.x = __expf(x.x - mx); x.y = __expf(x.y - mx);
      x.z = __expf(x.z - mx); x.w = __expf(x.w - mx);
      *(float4*)&srow[t] = x;
      sum += (x.x + x.y) + (x.z + x.w);
    }
    sum += __shfl_xor(sum, 16, 32);
    const float inv = 1.f / sum;
    float* arow = attn_out + (((size_t)b * NHEAD + h) * NND + (i0 + row)) * NND + jb;
    bf16_t* prow = &sP[row * NND + jb];
    for (int t = 0; t < 256; t += 4) {
      float4 x = *(const float4*)&srow[t];
      x.x *= inv; x.y *= inv; x.z *= inv; x.w *= inv;
      *(float4*)&arow[t] = x;                 // fp32 attn -> d_out
      PK4 pk;
      pk.h[0] = (bf16_t)x.x; pk.h[1] = (bf16_t)x.y;
      pk.h[2] = (bf16_t)x.z; pk.h[3] = (bf16_t)x.w;
      *(uint2*)&prow[t] = pk.u;               // bf16 for WMMA
    }
  }
  __syncthreads();

  // out[16,32] = attn[16,512] @ V[512,32]   (V stored transposed: [B,H,D,N])
#pragma unroll
  for (int n0 = 0; n0 < HDIM; n0 += 16) {
    v8f acc = {};
#pragma unroll
    for (int kt = 0; kt < 16; ++kt) {
      v16bf a2 = a_frag_g(&sP[kt * 32], NND, lane);
      const bf16_t* vp = vT + (((size_t)b * NHEAD + h) * HDIM + n0 + n) * NND +
                         kt * 32 + (lane & 16);
      acc = wmma_bf16(a2, b_frag_kcontig(vp), acc);
    }
#pragma unroll
    for (int r = 0; r < 8; ++r)
      ctx_bf[(size_t)(b * NND + i0 + r + mh) * HID + h * HDIM + n0 + n] = (bf16_t)acc[r];
  }
}

// ---------------- output projection + residual + LayerNorm + ReLU ----------------
__global__ __launch_bounds__(32) void k_final(const bf16_t* __restrict__ ctx_bf,
                                              const bf16_t* __restrict__ Wof,
                                              const float* __restrict__ bo,
                                              const float* __restrict__ node_h,
                                              const float* __restrict__ ln_g,
                                              const float* __restrict__ ln_b,
                                              float* __restrict__ out) {
  __shared__ __align__(16) float sY[16 * HID];
  const int lane = threadIdx.x;
  const int m0 = blockIdx.x * 16;
  const int n = lane & 15;
  const int mh = (lane & 16) >> 1;

#pragma unroll
  for (int nt = 0; nt < 16; ++nt) {
    v8f acc = {};
#pragma unroll
    for (int kt = 0; kt < 8; ++kt) {
      v16bf a = a_frag_g(ctx_bf + (size_t)m0 * HID + kt * 32, HID, lane);
      v16bf b = b_frag_packed(Wof + (size_t)(kt * 16 + nt) * 512, lane);
      acc = wmma_bf16(a, b, acc);
    }
#pragma unroll
    for (int r = 0; r < 8; ++r)
      sY[(r + mh) * HID + nt * 16 + n] = acc[r] + bo[nt * 16 + n];
  }
  __syncthreads();

  const int row = lane & 15;
  const int cb = (lane & 16) ? 128 : 0;
  const int gr = m0 + row;
  float sum = 0.f, sq = 0.f;
  for (int t = 0; t < 128; t += 4) {
    const float4 r4 = *(const float4*)&node_h[(size_t)gr * HID + cb + t];
    float4 y4 = *(const float4*)&sY[row * HID + cb + t];
    y4.x += r4.x; y4.y += r4.y; y4.z += r4.z; y4.w += r4.w;
    *(float4*)&sY[row * HID + cb + t] = y4;
    sum += (y4.x + y4.y) + (y4.z + y4.w);
    sq += (y4.x * y4.x + y4.y * y4.y) + (y4.z * y4.z + y4.w * y4.w);
  }
  sum += __shfl_xor(sum, 16, 32);
  sq  += __shfl_xor(sq, 16, 32);
  const float mu = sum * (1.f / HID);
  const float var = sq * (1.f / HID) - mu * mu;
  const float inv = rsqrtf(var + 1e-5f);
  for (int t = 0; t < 128; t += 4) {
    const float4 g4 = *(const float4*)&ln_g[cb + t];
    const float4 b4 = *(const float4*)&ln_b[cb + t];
    float4 y4 = *(const float4*)&sY[row * HID + cb + t];
    y4.x = fmaxf((y4.x - mu) * inv * g4.x + b4.x, 0.f);
    y4.y = fmaxf((y4.y - mu) * inv * g4.y + b4.y, 0.f);
    y4.z = fmaxf((y4.z - mu) * inv * g4.z + b4.z, 0.f);
    y4.w = fmaxf((y4.w - mu) * inv * g4.w + b4.w, 0.f);
    *(float4*)&out[(size_t)gr * HID + cb + t] = y4;
  }
}

// ---------------- launch ----------------
extern "C" void kernel_launch(void* const* d_in, const int* in_sizes, int n_in,
                              void* d_out, int out_size, void* d_ws, size_t ws_size,
                              hipStream_t stream) {
  (void)in_sizes; (void)n_in; (void)out_size; (void)ws_size;
  const float* node_h = (const float*)d_in[0];
  const float* edge_x = (const float*)d_in[1];
  const int*   emask  = (const int*)d_in[2];
  const float* Wq = (const float*)d_in[3];  const float* bq = (const float*)d_in[4];
  const float* Wk = (const float*)d_in[5];  const float* bk = (const float*)d_in[6];
  const float* Wv = (const float*)d_in[7];  const float* bv = (const float*)d_in[8];
  const float* We1 = (const float*)d_in[9];  const float* be1 = (const float*)d_in[10];
  const float* We2 = (const float*)d_in[11]; const float* be2 = (const float*)d_in[12];
  const float* Wo = (const float*)d_in[13]; const float* bo = (const float*)d_in[14];
  const float* ln_g = (const float*)d_in[15]; const float* ln_b = (const float*)d_in[16];

  float* out_x = (float*)d_out;
  float* out_attn = out_x + (size_t)BSZ * NND * HID;   // tuple element 2: attn [B,H,N,N]

  char* ws = (char*)d_ws;
  size_t off = 0;
  auto take = [&](size_t bytes) -> char* {
    char* p = ws + off;
    off += (bytes + 255) & ~(size_t)255;
    return p;
  };
  bf16_t* node_bf = (bf16_t*)take((size_t)BN * HID * 2);
  bf16_t* q_bf    = (bf16_t*)take((size_t)BN * HID * 2);
  bf16_t* kb_bf   = (bf16_t*)take((size_t)BN * HID * 2);
  bf16_t* vT      = (bf16_t*)take((size_t)BN * HID * 2);
  bf16_t* ctx_bf  = (bf16_t*)take((size_t)BN * HID * 2);
  bf16_t* Wqf     = (bf16_t*)take((size_t)HID * HID * 2);
  bf16_t* Wkf     = (bf16_t*)take((size_t)HID * HID * 2);
  bf16_t* Wvf     = (bf16_t*)take((size_t)HID * HID * 2);
  bf16_t* Wof     = (bf16_t*)take((size_t)HID * HID * 2);
  bf16_t* We1f    = (bf16_t*)take((size_t)32 * HID * 2);
  bf16_t* We2f    = (bf16_t*)take((size_t)HID * 16 * 2);
  float*  ebT     = (float*)take((size_t)BSZ * NHEAD * NND * NND * 4);

  // bf16 conversion of activations; fragment-order packing of all weights
  k_f32_to_bf16_v4<<<(BN * HID / 4 + 255) / 256, 256, 0, stream>>>(node_h, node_bf,
                                                                   BN * HID / 4);
  k_pack_frag<<<(HID * HID + 255) / 256, 256, 0, stream>>>(Wq, Wqf, 16, HID, HID, HID * HID);
  k_pack_frag<<<(HID * HID + 255) / 256, 256, 0, stream>>>(Wk, Wkf, 16, HID, HID, HID * HID);
  k_pack_frag<<<(HID * HID + 255) / 256, 256, 0, stream>>>(Wv, Wvf, 16, HID, HID, HID * HID);
  k_pack_frag<<<(HID * HID + 255) / 256, 256, 0, stream>>>(Wo, Wof, 16, HID, HID, HID * HID);
  k_pack_frag<<<(32 * HID + 255) / 256, 256, 0, stream>>>(We1, We1f, 16, EDIM, HID, 32 * HID);
  k_pack_frag<<<(HID * 16 + 255) / 256, 256, 0, stream>>>(We2, We2f, 1, HID, NHEAD, HID * 16);

  const dim3 w(32);
  const dim3 g1(HID / 16, BN / 16);
  k_gemm_bf16<<<g1, w, 0, stream>>>(node_bf, Wqf, bq, q_bf, 0);
  k_gemm_bf16<<<g1, w, 0, stream>>>(node_bf, Wkf, bk, kb_bf, 0);
  k_gemm_bf16<<<g1, w, 0, stream>>>(node_bf, Wvf, bv, vT, 1);

  k_edge_mlp<<<(BSZ * NND * NND) / 16, w, 0, stream>>>(edge_x, We1f, We2f, be1, be2, ebT);

  const dim3 ga(NND / 16, NHEAD, BSZ);
  k_attention<<<ga, w, 0, stream>>>(q_bf, kb_bf, vT, ebT, emask, out_attn, ctx_bf);

  k_final<<<BN / 16, w, 0, stream>>>(ctx_bf, Wof, bo, node_h, ln_g, ln_b, out_x);
}
